// GNN2_52123723104853
// MI455X (gfx1250) — matrix operand order repaired and checked by reading, not compiled, not run on hardware
//
#include <hip/hip_runtime.h>
#include <hip/hip_bf16.h>

// ---------------------------------------------------------------------------
// GCN stack for MI455X (gfx1250): 4 x [adj @ (h@W) + b -> ReLU -> BatchNorm]
// B=16, N=2048, C=128.
// Fast path: pre-convert adj/x/W to bf16 once; GEMM inner loops are pure
// global_load_b128 + v_wmma_f32_16x16x32_bf16 (fragment layouts fall out of
// bf16 row-major / transposed storage -> zero in-loop VALU conversion).
// Memory-bound by design: ~0.95 GB HBM total -> ~40us at 23.3 TB/s.
// Fallback path (small ws): round-1 convert-in-register kernels.
// ---------------------------------------------------------------------------

#define GB 16       // batch
#define GN 2048     // nodes
#define GC 128      // channels
#define GL 4        // layers
#define BN_EPS 1e-5f

typedef __attribute__((ext_vector_type(16))) __bf16          v16bf;
typedef __attribute__((ext_vector_type(8)))  float           v8f;
typedef __attribute__((ext_vector_type(16))) unsigned short  v16us;
typedef __attribute__((ext_vector_type(8)))  unsigned short  v8us;

__device__ __forceinline__ unsigned short f32_to_bf16_rne(float f) {
    unsigned int u = __float_as_uint(f);
    u += 0x7FFFu + ((u >> 16) & 1u);     // round-to-nearest-even
    return (unsigned short)(u >> 16);
}

// ---------------------------------------------------------------------------
// Conversion kernels (run once per launch).
// ---------------------------------------------------------------------------
__global__ __launch_bounds__(256) void cvt_f32_to_bf16(
    const float* __restrict__ in, unsigned short* __restrict__ out)
{
    const size_t i = ((size_t)blockIdx.x * 256 + threadIdx.x) * 8;
    float4 a = *(const float4*)(in + i);
    float4 b = *(const float4*)(in + i + 4);
    v8us o;
    o[0] = f32_to_bf16_rne(a.x); o[1] = f32_to_bf16_rne(a.y);
    o[2] = f32_to_bf16_rne(a.z); o[3] = f32_to_bf16_rne(a.w);
    o[4] = f32_to_bf16_rne(b.x); o[5] = f32_to_bf16_rne(b.y);
    o[6] = f32_to_bf16_rne(b.z); o[7] = f32_to_bf16_rne(b.w);
    *(v8us*)(out + i) = o;
}

// Wt[l][c][k] = bf16(W[l][k][c])  (transpose so B-fragments are contiguous)
__global__ __launch_bounds__(256) void cvt_w_transpose(
    const float* __restrict__ W, unsigned short* __restrict__ Wt)
{
    const int i = blockIdx.x * 256 + threadIdx.x;   // l*C*C + c*C + k
    const int l = i >> 14;
    const int c = (i >> 7) & (GC - 1);
    const int k = i & (GC - 1);
    Wt[i] = f32_to_bf16_rne(W[(size_t)l * GC * GC + (size_t)k * GC + c]);
}

// ---------------------------------------------------------------------------
// FAST Kernel 1: T = h @ W, h already bf16 [B,N,C], Wt bf16 [c][k].
// Wave = one 16x16 tile; A/B fragments loaded directly (contiguous K
// order): lane<16 row M=lane K 0..15; lane>=16 same row, K 16..31.
// Output transposed bf16: Tt[b][c][n].
// ---------------------------------------------------------------------------
__global__ __launch_bounds__(256) void gcn_xw_fast(
    const unsigned short* __restrict__ hB,   // [B,N,C] bf16
    const unsigned short* __restrict__ Wt,   // [C,C] bf16, [c][k]
    unsigned short* __restrict__ Tt)         // [B,C,N] bf16
{
    const int blk  = blockIdx.x;
    const int b    = blk >> 7;
    const int m0   = (blk & 127) << 4;
    const int tid  = threadIdx.x;
    const int lane = tid & 31;                // wave32
    const int wave = tid >> 5;
    const int c0   = wave << 4;
    const int half = lane >> 4;
    const int mrow = lane & 15;
    const int cc   = c0 + mrow;

    const unsigned short* aRow = hB + ((size_t)(b * GN + m0 + mrow)) * GC + half * 16;
    const unsigned short* bRow = Wt + (size_t)cc * GC + half * 16;

    v8f acc = {};
    #pragma unroll
    for (int kk = 0; kk < GC; kk += 32) {
        v8us a0 = *(const v8us*)(aRow + kk);        // K 0..7 of this half's 16
        v8us a1 = *(const v8us*)(aRow + kk + 8);    // K 8..15 of this half's 16
        v16us au;
        #pragma unroll
        for (int e = 0; e < 8; ++e) { au[e] = a0[e]; au[8 + e] = a1[e]; }
        v16us bu = *(const v16us*)(bRow + kk);      // contiguous K
        v16bf a  = __builtin_bit_cast(v16bf, au);
        v16bf bb = __builtin_bit_cast(v16bf, bu);
        acc = __builtin_amdgcn_wmma_f32_16x16x32_bf16(
            false, a, false, bb, (short)0, acc, false, false);
    }

    const int nbase = m0 + half * 8;
    v8us outv;
    #pragma unroll
    for (int r = 0; r < 8; ++r) outv[r] = f32_to_bf16_rne(acc[r]);
    *(v8us*)(Tt + ((size_t)(b * GC + cc)) * GN + nbase) = outv;
}

// ---------------------------------------------------------------------------
// FAST Kernel 2: Z = adjB @ T (bf16/bf16 -> f32), fused bias+ReLU+stats.
// No LDS in the K-loop: 4 b128 loads + 1 WMMA per step, unrolled.
// ---------------------------------------------------------------------------
__global__ __launch_bounds__(256) void gcn_adj_fast(
    const unsigned short* __restrict__ adjB, // [B,N,N] bf16
    const unsigned short* __restrict__ Tt,   // [B,C,N] bf16
    const float* __restrict__ bias,          // [C]
    float* __restrict__ htmp,                // [B,N,C] f32 pre-norm
    float* __restrict__ part)                // [2048, 2*C] block partials
{
    __shared__ float redS[256];
    __shared__ float redQ[256];

    const int blk  = blockIdx.x;
    const int b    = blk >> 7;
    const int m0   = (blk & 127) << 4;
    const int tid  = threadIdx.x;
    const int lane = tid & 31;
    const int wave = tid >> 5;
    const int c0   = wave << 4;
    const int half = lane >> 4;
    const int mrow = lane & 15;
    const int cc   = c0 + mrow;

    const unsigned short* aRow = adjB + ((size_t)(b * GN + m0 + mrow)) * GN + half * 16;
    const unsigned short* bRow = Tt + ((size_t)(b * GC + cc)) * GN + half * 16;

    v8f acc = {};
    #pragma unroll 4
    for (int kk = 0; kk < GN; kk += 32) {
        v8us a0 = *(const v8us*)(aRow + kk);
        v8us a1 = *(const v8us*)(aRow + kk + 8);
        v16us au;
        #pragma unroll
        for (int e = 0; e < 8; ++e) { au[e] = a0[e]; au[8 + e] = a1[e]; }
        v16us bu = *(const v16us*)(bRow + kk);
        if (kk + 256 < GN)                   // global_prefetch_b8, next lines
            __builtin_prefetch(aRow + kk + 256, 0, 3);
        v16bf a  = __builtin_bit_cast(v16bf, au);
        v16bf bb = __builtin_bit_cast(v16bf, bu);
        acc = __builtin_amdgcn_wmma_f32_16x16x32_bf16(
            false, a, false, bb, (short)0, acc, false, false);
    }

    // Fused epilogue: bias + ReLU, f32 activations for BN, local stats.
    const int nbase = m0 + half * 8;
    const float bv = bias[cc];
    float s = 0.f, q = 0.f;
    #pragma unroll
    for (int r = 0; r < 8; ++r) {
        float v = acc[r] + bv;
        v = fmaxf(v, 0.f);
        htmp[((size_t)(b * GN + nbase + r)) * GC + cc] = v;
        s += v;
        q += v * v;
    }
    // Deterministic in-block pairing: half0 lane adds its half1 partner.
    redS[tid] = s; redQ[tid] = q;
    __syncthreads();
    if ((tid & 16) == 0) {
        part[(size_t)blk * (2 * GC) + cc]      = redS[tid] + redS[tid + 16];
        part[(size_t)blk * (2 * GC) + GC + cc] = redQ[tid] + redQ[tid + 16];
    }
}

// ---------------------------------------------------------------------------
// FALLBACK kernels (small workspace): round-1 convert-in-register versions.
// ---------------------------------------------------------------------------
__global__ __launch_bounds__(256) void gcn_xw_f32(
    const float* __restrict__ h, const float* __restrict__ W,
    unsigned short* __restrict__ Tt)
{
    __shared__ float hTile[16 * 32];
    __shared__ float wTile[32 * GC];
    const int blk = blockIdx.x, b = blk >> 7, m0 = (blk & 127) << 4;
    const int tid = threadIdx.x, lane = tid & 31, wave = tid >> 5;
    const int c0 = wave << 4, half = lane >> 4, mrow = lane & 15;

    v8f acc = {};
    for (int kk = 0; kk < GC; kk += 32) {
        __syncthreads();
        {
            int i = tid * 2, r = i >> 5, c = i & 31;
            const float* src = h + ((size_t)(b * GN + m0 + r)) * GC + kk + c;
            hTile[i] = src[0]; hTile[i + 1] = src[1];
        }
        #pragma unroll
        for (int j = 0; j < 16; ++j) {
            int i = tid + j * 256;
            wTile[i] = W[(size_t)(kk + (i >> 7)) * GC + (i & (GC - 1))];
        }
        __syncthreads();
        v16us au, bu;
        #pragma unroll
        for (int e = 0; e < 16; ++e) {
            int k = e + half * 16;
            au[e] = f32_to_bf16_rne(hTile[mrow * 32 + k]);
        }
        #pragma unroll
        for (int e = 0; e < 16; ++e)
            bu[e] = f32_to_bf16_rne(wTile[(e + half * 16) * GC + c0 + mrow]);
        acc = __builtin_amdgcn_wmma_f32_16x16x32_bf16(
            false, __builtin_bit_cast(v16bf, au), false,
            __builtin_bit_cast(v16bf, bu), (short)0, acc, false, false);
    }
    const int cc = c0 + mrow, nbase = m0 + half * 8;
    v8us outv;
    #pragma unroll
    for (int r = 0; r < 8; ++r) outv[r] = f32_to_bf16_rne(acc[r]);
    *(v8us*)(Tt + ((size_t)(b * GC + cc)) * GN + nbase) = outv;
}

__global__ __launch_bounds__(256) void gcn_adj_f32(
    const float* __restrict__ adj, const unsigned short* __restrict__ Tt,
    const float* __restrict__ bias, float* __restrict__ htmp,
    float* __restrict__ part)
{
    __shared__ float adjTile[16 * 32];
    __shared__ float redS[256];
    __shared__ float redQ[256];
    const int blk = blockIdx.x, b = blk >> 7, m0 = (blk & 127) << 4;
    const int tid = threadIdx.x, lane = tid & 31, wave = tid >> 5;
    const int c0 = wave << 4, half = lane >> 4, mrow = lane & 15;
    const int cc = c0 + mrow;

    const size_t adjRow = (size_t)b * GN * GN + (size_t)m0 * GN;
    const unsigned short* tRow = Tt + ((size_t)(b * GC + cc)) * GN + half * 16;

    v8f acc = {};
    for (int kk = 0; kk < GN; kk += 32) {
        __syncthreads();
        {
            int i = tid * 2, r = i >> 5, c = i & 31;
            const float* src = adj + adjRow + (size_t)r * GN + kk + c;
            adjTile[i] = src[0]; adjTile[i + 1] = src[1];
        }
        __syncthreads();
        v16us au;
        #pragma unroll
        for (int e = 0; e < 16; ++e) {
            int k = e + half * 16;
            au[e] = f32_to_bf16_rne(adjTile[mrow * 32 + k]);
        }
        v16us bu = *(const v16us*)(tRow + kk);
        acc = __builtin_amdgcn_wmma_f32_16x16x32_bf16(
            false, __builtin_bit_cast(v16bf, au), false,
            __builtin_bit_cast(v16bf, bu), (short)0, acc, false, false);
    }
    const int nbase = m0 + half * 8;
    const float bv = bias[cc];
    float s = 0.f, q = 0.f;
    #pragma unroll
    for (int r = 0; r < 8; ++r) {
        float v = fmaxf(acc[r] + bv, 0.f);
        htmp[((size_t)(b * GN + nbase + r)) * GC + cc] = v;
        s += v; q += v * v;
    }
    redS[tid] = s; redQ[tid] = q;
    __syncthreads();
    if ((tid & 16) == 0) {
        part[(size_t)blk * (2 * GC) + cc]      = redS[tid] + redS[tid + 16];
        part[(size_t)blk * (2 * GC) + GC + cc] = redQ[tid] + redQ[tid + 16];
    }
}

// ---------------------------------------------------------------------------
// Deterministic tree-reduce of block partials -> stats[2*C].
// ---------------------------------------------------------------------------
__global__ __launch_bounds__(256) void gcn_reduce_kernel(
    const float* __restrict__ part, float* __restrict__ stats)
{
    __shared__ float red[256];
    const int e = blockIdx.x, tid = threadIdx.x;
    float s = 0.f;
    for (int blk = tid; blk < GB * (GN / 16); blk += 256)
        s += part[(size_t)blk * (2 * GC) + e];
    red[tid] = s;
    __syncthreads();
    #pragma unroll
    for (int off = 128; off > 0; off >>= 1) {
        if (tid < off) red[tid] += red[tid + off];
        __syncthreads();
    }
    if (tid == 0) stats[e] = red[0];
}

// ---------------------------------------------------------------------------
// BatchNorm apply. Writes f32 (outF) and/or bf16 (outB) as requested.
// ---------------------------------------------------------------------------
__global__ __launch_bounds__(256) void gcn_bn_kernel(
    const float* __restrict__ htmp, const float* __restrict__ stats,
    const float* __restrict__ gamma, const float* __restrict__ beta,
    float* __restrict__ outF, unsigned short* __restrict__ outB)
{
    const size_t idx = ((size_t)blockIdx.x * 256 + threadIdx.x) * 4;
    const int c = (int)(idx & (GC - 1));
    const float invCnt = 1.0f / (float)(GB * GN);
    float4 hv = *(const float4*)(htmp + idx);
    const float hvv[4] = { hv.x, hv.y, hv.z, hv.w };
    float o[4];
    #pragma unroll
    for (int j = 0; j < 4; ++j) {
        float mu  = stats[c + j] * invCnt;
        float var = stats[GC + c + j] * invCnt - mu * mu;
        float inv = rsqrtf(var + BN_EPS);
        o[j] = gamma[c + j] * (hvv[j] - mu) * inv + beta[c + j];
    }
    if (outF) *(float4*)(outF + idx) = make_float4(o[0], o[1], o[2], o[3]);
    if (outB) {
        v8us tmp;  // use low 4 lanes only
        tmp[0] = f32_to_bf16_rne(o[0]); tmp[1] = f32_to_bf16_rne(o[1]);
        tmp[2] = f32_to_bf16_rne(o[2]); tmp[3] = f32_to_bf16_rne(o[3]);
        unsigned long long packed =
            (unsigned long long)tmp[0] | ((unsigned long long)tmp[1] << 16) |
            ((unsigned long long)tmp[2] << 32) | ((unsigned long long)tmp[3] << 48);
        *(unsigned long long*)(outB + idx) = packed;
    }
}

// ---------------------------------------------------------------------------
// Host-side launch. Inputs: x, adj, W, b, gamma, beta (all f32).
// Fast ws (~178 MB): adjB | xB | hbufB | Tt | htmp | Wt | part | stats
// Fallback ws (~42 MB): Tt | htmp | hbufF | part | stats
// ---------------------------------------------------------------------------
extern "C" void kernel_launch(void* const* d_in, const int* in_sizes, int n_in,
                              void* d_out, int out_size, void* d_ws, size_t ws_size,
                              hipStream_t stream) {
    (void)in_sizes; (void)n_in; (void)out_size;
    const float* x     = (const float*)d_in[0];
    const float* adj   = (const float*)d_in[1];
    const float* W     = (const float*)d_in[2];
    const float* bias  = (const float*)d_in[3];
    const float* gamma = (const float*)d_in[4];
    const float* beta  = (const float*)d_in[5];
    float* outp = (float*)d_out;
    char* ws = (char*)d_ws;

    const size_t ADJ_N   = (size_t)GB * GN * GN;            // 67,108,864
    const size_t X_N     = (size_t)GB * GN * GC;            //  4,194,304
    const size_t ADJB_B  = ADJ_N * 2;                       // 134 MB
    const size_t XB_B    = X_N * 2;                         //   8 MB
    const size_t TT_B    = X_N * 2;                         //   8 MB
    const size_t HT_B    = X_N * 4;                         //  16 MB
    const size_t WT_B    = (size_t)GL * GC * GC * 2;        // 128 KB
    const size_t NBLK    = (size_t)GB * (GN / 16);          // 2048
    const size_t PART_B  = NBLK * 2 * GC * 4;               //   2 MB
    const size_t STAT_B  = (size_t)GL * 2 * GC * 4;         //   4 KB
    const size_t fastNeed = ADJB_B + XB_B + TT_B + HT_B + XB_B + WT_B + PART_B + STAT_B;

    const int gemmBlocks = (int)NBLK;                       // 2048
    const int bnBlocks   = (int)(X_N / (256 * 4));          // 4096

    if (ws_size >= fastNeed) {
        // ---- fast path: one-time bf16 conversions, then conversion-free GEMMs
        unsigned short* adjB  = (unsigned short*)ws;
        unsigned short* xB    = (unsigned short*)(ws + ADJB_B);
        unsigned short* hbufB = (unsigned short*)(ws + ADJB_B + XB_B);
        unsigned short* Tt    = (unsigned short*)(ws + ADJB_B + 2 * XB_B);
        float*          htmp  = (float*)(ws + ADJB_B + 2 * XB_B + TT_B);
        unsigned short* Wt    = (unsigned short*)(ws + ADJB_B + 2 * XB_B + TT_B + HT_B);
        float*          partb = (float*)(ws + ADJB_B + 2 * XB_B + TT_B + HT_B + WT_B);
        float*          stats = (float*)(ws + ADJB_B + 2 * XB_B + TT_B + HT_B + WT_B + PART_B);

        cvt_f32_to_bf16<<<(int)(ADJ_N / (256 * 8)), 256, 0, stream>>>(adj, adjB);
        cvt_f32_to_bf16<<<(int)(X_N / (256 * 8)), 256, 0, stream>>>(x, xB);
        cvt_w_transpose<<<(int)(GL * GC * GC / 256), 256, 0, stream>>>(W, Wt);

        const unsigned short* hin = xB;
        for (int l = 0; l < GL; ++l) {
            gcn_xw_fast<<<gemmBlocks, 256, 0, stream>>>(
                hin, Wt + (size_t)l * GC * GC, Tt);
            gcn_adj_fast<<<gemmBlocks, 256, 0, stream>>>(
                adjB, Tt, bias + (size_t)l * GC, htmp, partb);
            gcn_reduce_kernel<<<2 * GC, 256, 0, stream>>>(
                partb, stats + (size_t)l * 2 * GC);
            const bool last = (l == GL - 1);
            gcn_bn_kernel<<<bnBlocks, 256, 0, stream>>>(
                htmp, stats + (size_t)l * 2 * GC,
                gamma + (size_t)l * GC, beta + (size_t)l * GC,
                last ? outp : nullptr, last ? nullptr : hbufB);
            hin = hbufB;
        }
    } else {
        // ---- fallback path: f32 inputs, convert-in-register (round-1 kernels)
        unsigned short* Tt    = (unsigned short*)ws;
        float*          htmp  = (float*)(ws + TT_B);
        float*          hbufF = (float*)(ws + TT_B + HT_B);
        float*          partb = (float*)(ws + TT_B + 2 * HT_B);
        float*          stats = (float*)(ws + TT_B + 2 * HT_B + PART_B);

        const float* hin = x;
        for (int l = 0; l < GL; ++l) {
            gcn_xw_f32<<<gemmBlocks, 256, 0, stream>>>(
                hin, W + (size_t)l * GC * GC, Tt);
            gcn_adj_f32<<<gemmBlocks, 256, 0, stream>>>(
                adj, Tt, bias + (size_t)l * GC, htmp, partb);
            gcn_reduce_kernel<<<2 * GC, 256, 0, stream>>>(
                partb, stats + (size_t)l * 2 * GC);
            const bool last = (l == GL - 1);
            gcn_bn_kernel<<<bnBlocks, 256, 0, stream>>>(
                htmp, stats + (size_t)l * 2 * GC,
                gamma + (size_t)l * GC, beta + (size_t)l * GC,
                last ? outp : hbufF, nullptr);
            hin = hbufF;
        }
    }
}
